// SimpleMoE_Loop_85770496901556
// MI455X (gfx1250) — matrix operand order repaired and checked by reading, not compiled, not run on hardware
//
#include <hip/hip_runtime.h>
#include <hip/hip_bf16.h>

#define N_TOKENS 16384
#define H_DIM    1024
#define FFN_DIM  4096
#define NE       8
#define MAX_SLOTS (N_TOKENS * 2)

#if __has_builtin(__builtin_amdgcn_global_load_async_to_lds_b128) && \
    __has_builtin(__builtin_amdgcn_s_wait_asynccnt)
#define USE_ASYNC 1
#else
#define USE_ASYNC 0
#endif

#if __has_builtin(__builtin_amdgcn_tensor_load_to_lds) && \
    __has_builtin(__builtin_amdgcn_s_wait_tensorcnt)
#define USE_TDM 1
#else
#define USE_TDM 0
#endif

typedef __attribute__((ext_vector_type(8)))  float  v8f;
typedef __attribute__((ext_vector_type(16))) __bf16 v16bf;
typedef int v4i_ __attribute__((vector_size(16)));
typedef __attribute__((address_space(1))) v4i_ gv4i;
typedef __attribute__((address_space(3))) v4i_ lv4i;
typedef __attribute__((ext_vector_type(4))) unsigned int u32x4;
typedef __attribute__((ext_vector_type(8))) int i32x8;
typedef __attribute__((ext_vector_type(4))) int i32x4;

union FragBF {
    v16bf v;
    uint4 q[2];
};

__device__ __forceinline__ unsigned short f32_to_bf16(float f) {
    union { float f; unsigned int u; } cv;
    cv.f = f;
    unsigned int u = cv.u;
    u += 0x7FFFu + ((u >> 16) & 1u);   // round-to-nearest-even
    return (unsigned short)(u >> 16);
}

// Fill one 32-halfword A row (64B) into LDS: async (ASYNCcnt path) or sync.
__device__ __forceinline__ void fill_a_row(unsigned short* dst,
                                           const unsigned short* src) {
#if USE_ASYNC
#pragma unroll
    for (int i = 0; i < 4; ++i)
        __builtin_amdgcn_global_load_async_to_lds_b128(
            (gv4i*)(const_cast<unsigned short*>(src) + i * 8),
            (lv4i*)(dst + i * 8),
            0, 0);
#else
    const uint4* s = (const uint4*)src;
    uint4* d = (uint4*)dst;
#pragma unroll
    for (int i = 0; i < 4; ++i) d[i] = s[i];
#endif
}

__device__ __forceinline__ void wait_async_fills() {
#if USE_ASYNC
    __builtin_amdgcn_s_wait_asynccnt(0);
#endif
}

__device__ __forceinline__ unsigned lds_byte_addr(const void* p) {
    return (unsigned)(unsigned long long)(__attribute__((address_space(3))) const void*)p;
}

#if USE_TDM
// TDM: load a dense 2D bf16 tile (tile_dim1 rows x 32 halves) from global into
// LDS, padding 4 dwords after every 16 dwords -> LDS row stride 40 halves.
// Rows >= rows_valid read as zero (tensor_dim1 OOB).  One issue per wave.
__device__ __forceinline__ void tdm_load_a_tile(unsigned lds_addr,
                                                const void* gaddr,
                                                unsigned rows_valid,
                                                unsigned row_stride_elems) {
    unsigned long long ga = (unsigned long long)gaddr;
    u32x4 g0;
    g0[0] = 1u;                                            // count=1, user mode
    g0[1] = lds_addr;                                      // LDS dest (bytes)
    g0[2] = (unsigned)(ga & 0xFFFFFFFFu);                  // global_addr[31:0]
    g0[3] = (unsigned)((ga >> 32) & 0x1FFFFFFu)            // global_addr[56:32]
          | 0x80000000u;                                   // type=2 (image)
    i32x8 g1;
    g1[0] = (int)(0x10000u          // data_size=1 -> 2-byte elements
          | 0x100000u               // pad_enable
          | (3u << 22)              // pad_interval: 16 dwords
          | (3u << 25));            // pad_amount: 4 dwords
    g1[1] = (int)(32u << 16);                              // tensor_dim0 = 32
    g1[2] = (int)((rows_valid & 0xFFFFu) << 16);           // tensor_dim1[15:0]
    g1[3] = (int)(((rows_valid >> 16) & 0xFFFFu)           // tensor_dim1[31:16]
          | (32u << 16));                                  // tile_dim0 = 32
    g1[4] = (int)128u;                                     // tile_dim1=128, tile_dim2=0
    g1[5] = (int)row_stride_elems;                         // tensor_dim0_stride lo
    g1[6] = 0;                                             // stride hi, dim1_stride lo
    g1[7] = 0;                                             // dim1_stride hi
    __builtin_amdgcn_tensor_load_to_lds(g0, g1, (i32x4)(0), (i32x4)(0),
                                        (i32x8)(0), 0);
}
#endif

// ---------------------------------------------------------------------------
// fp32 -> bf16 bulk convert
// ---------------------------------------------------------------------------
__global__ void convert_bf16_kernel(const float* __restrict__ src,
                                    unsigned short* __restrict__ dst,
                                    long long n4) {
    long long i = (long long)blockIdx.x * blockDim.x + threadIdx.x;
    long long stride = (long long)gridDim.x * blockDim.x;
    for (; i < n4; i += stride) {
        float4 v = ((const float4*)src)[i];
        unsigned int lo = (unsigned int)f32_to_bf16(v.x) |
                          ((unsigned int)f32_to_bf16(v.y) << 16);
        unsigned int hi = (unsigned int)f32_to_bf16(v.z) |
                          ((unsigned int)f32_to_bf16(v.w) << 16);
        uint2 o; o.x = lo; o.y = hi;
        ((uint2*)dst)[i] = o;
    }
}

__global__ void init_kernel(int* __restrict__ counts, int* __restrict__ cursors) {
    int t = threadIdx.x;
    if (t < NE) { counts[t] = 0; cursors[t] = 0; }
}

// ---------------------------------------------------------------------------
// gate: one wave32 per token
// ---------------------------------------------------------------------------
__global__ __launch_bounds__(256) void gate_kernel(
        const float* __restrict__ x, const float* __restrict__ Wg,
        const float* __restrict__ bg,
        int2* __restrict__ tokE, float2* __restrict__ tokW,
        int* __restrict__ counts) {
    const int token = blockIdx.x * 8 + (threadIdx.x >> 5);
    const int lane  = threadIdx.x & 31;

    float acc[NE];
#pragma unroll
    for (int e = 0; e < NE; ++e) acc[e] = 0.0f;

    const float* xr = x + (long long)token * H_DIM;
    for (int h = lane; h < H_DIM; h += 32) {
        float xv = xr[h];
        const float4* wr = (const float4*)(Wg + h * NE);
        float4 w0 = wr[0], w1 = wr[1];
        acc[0] += xv * w0.x; acc[1] += xv * w0.y;
        acc[2] += xv * w0.z; acc[3] += xv * w0.w;
        acc[4] += xv * w1.x; acc[5] += xv * w1.y;
        acc[6] += xv * w1.z; acc[7] += xv * w1.w;
    }
#pragma unroll
    for (int o = 16; o > 0; o >>= 1) {
#pragma unroll
        for (int e = 0; e < NE; ++e)
            acc[e] += __shfl_xor(acc[e], o, 32);
    }

    if (lane == 0) {
#pragma unroll
        for (int e = 0; e < NE; ++e) acc[e] += bg[e];
        int   i0 = 0;       float s0 = acc[0];
#pragma unroll
        for (int e = 1; e < NE; ++e) { if (acc[e] > s0) { s0 = acc[e]; i0 = e; } }
        int   i1 = -1;      float s1 = -3.4e38f;
#pragma unroll
        for (int e = 0; e < NE; ++e) { if (e != i0 && acc[e] > s1) { s1 = acc[e]; i1 = e; } }
        float w0 = 1.0f / (1.0f + __expf(s1 - s0));
        float w1 = 1.0f - w0;
        atomicAdd(&counts[i0], 1);
        atomicAdd(&counts[i1], 1);
        tokE[token] = make_int2(i0, i1);
        tokW[token] = make_float2(w0, w1);
    }
}

__global__ void scan_kernel(const int* __restrict__ counts, int* __restrict__ offsets) {
    if (threadIdx.x == 0) {
        int run = 0;
        for (int e = 0; e < NE; ++e) { offsets[e] = run; run += counts[e]; }
    }
}

__global__ __launch_bounds__(256) void scatter_kernel(
        const int2* __restrict__ tokE, const float2* __restrict__ tokW,
        const int* __restrict__ offsets, int* __restrict__ cursors,
        int* __restrict__ slotTok, float* __restrict__ slotW) {
    const int token = blockIdx.x * blockDim.x + threadIdx.x;
    if (token >= N_TOKENS) return;
    int2   ee = tokE[token];
    float2 ww = tokW[token];
    int p0 = atomicAdd(&cursors[ee.x], 1);
    int g0 = offsets[ee.x] + p0;
    slotTok[g0] = token; slotW[g0] = ww.x;
    int p1 = atomicAdd(&cursors[ee.y], 1);
    int g1 = offsets[ee.y] + p1;
    slotTok[g1] = token; slotW[g1] = ww.y;
}

// ---------------------------------------------------------------------------
// FFN1: h = relu(gather(x) @ W1[e] + b1[e])
// Gathered A rows -> per-lane async global->LDS fills, double buffered.
// ---------------------------------------------------------------------------
__global__ __launch_bounds__(128) void ffn1_kernel(
        const unsigned short* __restrict__ xb,
        const unsigned short* __restrict__ w1b,
        const float* __restrict__ b1,
        const int* __restrict__ slotTok,
        const int* __restrict__ counts,
        const int* __restrict__ offsets,
        unsigned short* __restrict__ hbuf) {
    const int e     = blockIdx.z;
    const int cnt   = counts[e];
    const int mbase = blockIdx.x * 128;
    if (mbase >= cnt) return;
    const int off = offsets[e];
    const int nb  = blockIdx.y * 128;

    __shared__ unsigned short As[2][128][40];   // [buf][m][k] (+8 pad halves)
    __shared__ unsigned short Bs[2][128][40];   // transposed: [buf][n][k]

    const int tid  = threadIdx.x;
    const int lane = tid & 31;
    const int wave = tid >> 5;
    const int wm   = (wave >> 1) * 64;
    const int wn   = (wave & 1) * 64;

    const bool myValid = (mbase + tid) < cnt;
    int myTok = 0;
    if (myValid) myTok = slotTok[off + mbase + tid];
    const unsigned short* myRow = xb + (long long)myTok * H_DIM;

    const long long w1base = (long long)e * H_DIM * FFN_DIM + nb;

    int bkr[4], bnc[4];
#pragma unroll
    for (int i = 0; i < 4; ++i) {
        int q = i * 128 + tid;
        bkr[i] = q >> 4;
        bnc[i] = (q & 15) * 8;
    }

    if (myValid) {
        fill_a_row(&As[0][tid][0], myRow);
    } else {
        uint4 z = make_uint4(0u, 0u, 0u, 0u);
#pragma unroll
        for (int b = 0; b < 2; ++b) {
            *(uint4*)&As[b][tid][0] = z;  *(uint4*)&As[b][tid][8] = z;
            *(uint4*)&As[b][tid][16] = z; *(uint4*)&As[b][tid][24] = z;
        }
    }
    uint4 breg[4];
#pragma unroll
    for (int i = 0; i < 4; ++i)
        breg[i] = *(const uint4*)(w1b + w1base + (long long)bkr[i] * FFN_DIM + bnc[i]);

    v8f acc[4][4] = {};

    int cur = 0;
    for (int kc = 0; kc < H_DIM; kc += 32) {
        const int nxt = cur ^ 1;
#pragma unroll
        for (int i = 0; i < 4; ++i) {
            const unsigned short* hs = (const unsigned short*)&breg[i];
#pragma unroll
            for (int j = 0; j < 8; ++j) Bs[cur][bnc[i] + j][bkr[i]] = hs[j];
        }
        wait_async_fills();
        __syncthreads();

        const bool more = (kc + 32) < H_DIM;
        if (more) {
            if (myValid) fill_a_row(&As[nxt][tid][0], myRow + kc + 32);
#pragma unroll
            for (int i = 0; i < 4; ++i)
                breg[i] = *(const uint4*)(w1b + w1base +
                          (long long)(kc + 32 + bkr[i]) * FFN_DIM + bnc[i]);
            __builtin_prefetch(w1b + w1base + (long long)(kc + 64 + (tid >> 2)) * FFN_DIM +
                               (tid & 3) * 32, 0, 0);
        }

        const int hh  = lane >> 4;
        const int r15 = lane & 15;
        FragBF a[4], b[4];
#pragma unroll
        for (int mi = 0; mi < 4; ++mi) {
            int row = wm + mi * 16 + r15;
            a[mi].q[0] = *(const uint4*)&As[cur][row][hh * 8];
            a[mi].q[1] = *(const uint4*)&As[cur][row][16 + hh * 8];
        }
#pragma unroll
        for (int ni = 0; ni < 4; ++ni) {
            int col = wn + ni * 16 + r15;
            b[ni].q[0] = *(const uint4*)&Bs[cur][col][hh * 8];
            b[ni].q[1] = *(const uint4*)&Bs[cur][col][16 + hh * 8];
        }
#pragma unroll
        for (int mi = 0; mi < 4; ++mi)
#pragma unroll
            for (int ni = 0; ni < 4; ++ni)
                acc[mi][ni] = __builtin_amdgcn_wmma_f32_16x16x32_bf16(
                    false, a[mi].v, false, b[ni].v, (short)0, acc[mi][ni], false, false);
        __syncthreads();
        cur = nxt;
    }

    const int hh  = lane >> 4;
    const int r15 = lane & 15;
#pragma unroll
    for (int ni = 0; ni < 4; ++ni) {
        int nglob  = nb + wn + ni * 16 + r15;
        float bias = b1[e * FFN_DIM + nglob];
#pragma unroll
        for (int mi = 0; mi < 4; ++mi) {
#pragma unroll
            for (int v = 0; v < 8; ++v) {
                int m    = wm + mi * 16 + v + hh * 8;
                int srow = mbase + m;
                if (srow < cnt) {
                    float val = acc[mi][ni][v] + bias;
                    val = val > 0.0f ? val : 0.0f;
                    hbuf[(long long)(off + srow) * FFN_DIM + nglob] = f32_to_bf16(val);
                }
            }
        }
    }
}

// ---------------------------------------------------------------------------
// FFN2: y = h @ W2[e] + b2[e]; out[token] += w * y
// Dense contiguous A rows -> one TDM tensor_load_to_lds per tile (wave 0),
// padded to the LDS fragment stride; rows past cnt zero-filled via OOB.
// ---------------------------------------------------------------------------
__global__ __launch_bounds__(128) void ffn2_kernel(
        const unsigned short* __restrict__ hbuf,
        const unsigned short* __restrict__ w2b,
        const float* __restrict__ b2,
        const int* __restrict__ slotTok,
        const float* __restrict__ slotW,
        const int* __restrict__ counts,
        const int* __restrict__ offsets,
        float* __restrict__ out) {
    const int e     = blockIdx.z;
    const int cnt   = counts[e];
    const int mbase = blockIdx.x * 128;
    if (mbase >= cnt) return;
    const int off = offsets[e];
    const int nb  = blockIdx.y * 128;

    __shared__ unsigned short As[2][128][40];
    __shared__ unsigned short Bs[2][128][40];
    __shared__ int   tokS[128];
    __shared__ float wS[128];

    const int tid  = threadIdx.x;
    const int lane = tid & 31;
    const int wave = tid >> 5;
    const int wm   = (wave >> 1) * 64;
    const int wn   = (wave & 1) * 64;

    const bool myValid = (mbase + tid) < cnt;
    const long long myRowIdx = (long long)(off + mbase + tid);
    tokS[tid] = myValid ? slotTok[myRowIdx] : 0;
    wS[tid]   = myValid ? slotW[myRowIdx]   : 0.0f;

    const unsigned rowsValid = (unsigned)(cnt - mbase);      // > 0 here
    const unsigned short* tileBase = hbuf + (long long)(off + mbase) * FFN_DIM;

    const long long w2base = (long long)e * FFN_DIM * H_DIM + nb;

    int bkr[4], bnc[4];
#pragma unroll
    for (int i = 0; i < 4; ++i) {
        int q = i * 128 + tid;
        bkr[i] = q >> 4;
        bnc[i] = (q & 15) * 8;
    }

    // ---- prologue: A tile 0
#if USE_TDM
    if (wave == 0)
        tdm_load_a_tile(lds_byte_addr(&As[0][0][0]), tileBase, rowsValid, FFN_DIM);
#else
    if (myValid) {
        fill_a_row(&As[0][tid][0], hbuf + myRowIdx * FFN_DIM);
    } else {
        uint4 z = make_uint4(0u, 0u, 0u, 0u);
#pragma unroll
        for (int b = 0; b < 2; ++b) {
            *(uint4*)&As[b][tid][0] = z;  *(uint4*)&As[b][tid][8] = z;
            *(uint4*)&As[b][tid][16] = z; *(uint4*)&As[b][tid][24] = z;
        }
    }
#endif
    uint4 breg[4];
#pragma unroll
    for (int i = 0; i < 4; ++i)
        breg[i] = *(const uint4*)(w2b + w2base + (long long)bkr[i] * H_DIM + bnc[i]);

    v8f acc[4][4] = {};

    int cur = 0;
    for (int kc = 0; kc < FFN_DIM; kc += 32) {
        const int nxt = cur ^ 1;
#pragma unroll
        for (int i = 0; i < 4; ++i) {
            const unsigned short* hs = (const unsigned short*)&breg[i];
#pragma unroll
            for (int j = 0; j < 8; ++j) Bs[cur][bnc[i] + j][bkr[i]] = hs[j];
        }
#if USE_TDM
        if (wave == 0) __builtin_amdgcn_s_wait_tensorcnt(0);
#else
        wait_async_fills();
#endif
        __syncthreads();

        const bool more = (kc + 32) < FFN_DIM;
        if (more) {
#if USE_TDM
            if (wave == 0)
                tdm_load_a_tile(lds_byte_addr(&As[nxt][0][0]),
                                tileBase + kc + 32, rowsValid, FFN_DIM);
#else
            if (myValid) fill_a_row(&As[nxt][tid][0], hbuf + myRowIdx * FFN_DIM + kc + 32);
#endif
#pragma unroll
            for (int i = 0; i < 4; ++i)
                breg[i] = *(const uint4*)(w2b + w2base +
                          (long long)(kc + 32 + bkr[i]) * H_DIM + bnc[i]);
            __builtin_prefetch(w2b + w2base + (long long)(kc + 64 + (tid >> 2)) * H_DIM +
                               (tid & 3) * 32, 0, 0);
        }

        const int hh  = lane >> 4;
        const int r15 = lane & 15;
        FragBF a[4], b[4];
#pragma unroll
        for (int mi = 0; mi < 4; ++mi) {
            int row = wm + mi * 16 + r15;
            a[mi].q[0] = *(const uint4*)&As[cur][row][hh * 8];
            a[mi].q[1] = *(const uint4*)&As[cur][row][16 + hh * 8];
        }
#pragma unroll
        for (int ni = 0; ni < 4; ++ni) {
            int col = wn + ni * 16 + r15;
            b[ni].q[0] = *(const uint4*)&Bs[cur][col][hh * 8];
            b[ni].q[1] = *(const uint4*)&Bs[cur][col][16 + hh * 8];
        }
#pragma unroll
        for (int mi = 0; mi < 4; ++mi)
#pragma unroll
            for (int ni = 0; ni < 4; ++ni)
                acc[mi][ni] = __builtin_amdgcn_wmma_f32_16x16x32_bf16(
                    false, a[mi].v, false, b[ni].v, (short)0, acc[mi][ni], false, false);
        __syncthreads();
        cur = nxt;
    }

    const int hh  = lane >> 4;
    const int r15 = lane & 15;
#pragma unroll
    for (int ni = 0; ni < 4; ++ni) {
        int nglob  = nb + wn + ni * 16 + r15;
        float bias = b2[e * H_DIM + nglob];
#pragma unroll
        for (int mi = 0; mi < 4; ++mi) {
#pragma unroll
            for (int v = 0; v < 8; ++v) {
                int m    = wm + mi * 16 + v + hh * 8;
                int srow = mbase + m;
                if (srow < cnt) {
                    float y = acc[mi][ni][v] + bias;
                    atomicAdd(&out[(long long)tokS[m] * H_DIM + nglob], wS[m] * y);
                }
            }
        }
    }
}

// ---------------------------------------------------------------------------
// launcher
// ---------------------------------------------------------------------------
extern "C" void kernel_launch(void* const* d_in, const int* in_sizes, int n_in,
                              void* d_out, int out_size, void* d_ws, size_t ws_size,
                              hipStream_t stream) {
    (void)in_sizes; (void)n_in; (void)out_size; (void)ws_size;

    const float* x  = (const float*)d_in[0];
    const float* Wg = (const float*)d_in[1];
    const float* bg = (const float*)d_in[2];
    const float* W1 = (const float*)d_in[3];
    const float* b1 = (const float*)d_in[4];
    const float* W2 = (const float*)d_in[5];
    const float* b2 = (const float*)d_in[6];
    float* out = (float*)d_out;

    char* ws = (char*)d_ws;
    size_t o = 0;
    auto carve = [&](size_t bytes) -> void* {
        void* p = ws + o;
        o = (o + bytes + 255) & ~(size_t)255;
        return p;
    };
    int*            counts  = (int*)carve(NE * sizeof(int));
    int*            cursors = (int*)carve(NE * sizeof(int));
    int*            offsets = (int*)carve(NE * sizeof(int));
    int2*           tokE    = (int2*)carve(N_TOKENS * sizeof(int2));
    float2*         tokW    = (float2*)carve(N_TOKENS * sizeof(float2));
    int*            slotTok = (int*)carve(MAX_SLOTS * sizeof(int));
    float*          slotW   = (float*)carve(MAX_SLOTS * sizeof(float));
    unsigned short* xb      = (unsigned short*)carve((size_t)N_TOKENS * H_DIM * 2);
    unsigned short* w1b     = (unsigned short*)carve((size_t)NE * H_DIM * FFN_DIM * 2);
    unsigned short* w2b     = (unsigned short*)carve((size_t)NE * FFN_DIM * H_DIM * 2);
    unsigned short* hbuf    = (unsigned short*)carve((size_t)MAX_SLOTS * FFN_DIM * 2);

    (void)hipMemsetAsync(d_out, 0, (size_t)N_TOKENS * H_DIM * sizeof(float), stream);

    init_kernel<<<1, 32, 0, stream>>>(counts, cursors);

    convert_bf16_kernel<<<2048, 256, 0, stream>>>(x,  xb,  (long long)N_TOKENS * H_DIM / 4);
    convert_bf16_kernel<<<4096, 256, 0, stream>>>(W1, w1b, (long long)NE * H_DIM * FFN_DIM / 4);
    convert_bf16_kernel<<<4096, 256, 0, stream>>>(W2, w2b, (long long)NE * FFN_DIM * H_DIM / 4);

    gate_kernel<<<N_TOKENS / 8, 256, 0, stream>>>(x, Wg, bg, tokE, tokW, counts);
    scan_kernel<<<1, 1, 0, stream>>>(counts, offsets);
    scatter_kernel<<<N_TOKENS / 256, 256, 0, stream>>>(tokE, tokW, offsets, cursors,
                                                       slotTok, slotW);

    dim3 g1(N_TOKENS / 128, FFN_DIM / 128, NE);
    ffn1_kernel<<<g1, 128, 0, stream>>>(xb, w1b, b1, slotTok, counts, offsets, hbuf);

    dim3 g2(N_TOKENS / 128, H_DIM / 128, NE);
    ffn2_kernel<<<g2, 128, 0, stream>>>(hbuf, w2b, b2, slotTok, slotW, counts, offsets, out);
}
